// MaskedCausalAttention_40827959116101
// MI455X (gfx1250) — matrix operand order repaired and checked by reading, not compiled, not run on hardware
//
#include <hip/hip_runtime.h>
#include <hip/hip_bf16.h>

#define Tq 2048
#define Dh 128
#define NH 16
#define NB 2

typedef __attribute__((ext_vector_type(16))) _Float16 v16h;
typedef __attribute__((ext_vector_type(8)))  _Float16 v8h;
typedef __attribute__((ext_vector_type(8)))  float    v8f;

__device__ __forceinline__ v16h make16(v8h lo, v8h hi) {
    v16h r;
#pragma unroll
    for (int i = 0; i < 8; ++i) { r[i] = lo[i]; r[i + 8] = hi[i]; }
    return r;
}

__device__ __forceinline__ v16h ld_frag(const _Float16* p0, const _Float16* p1) {
    v8h lo = *(const v8h*)p0;
    v8h hi = *(const v8h*)p1;
    return make16(lo, hi);
}

__device__ __forceinline__ v8f zero8() {
    v8f r;
#pragma unroll
    for (int i = 0; i < 8; ++i) r[i] = 0.0f;
    return r;
}

// ---------------------------------------------------------------------------
// One projection panel: stage W (128x128 f32 -> f16 LDS), 32 WMMAs per wave,
// straight-line store.  TRANSV=false: f16 head-major [(head)*T + t]*128 + e.
// TRANSV=true:  f16 V^T layout [(head)*128 + e]*T + t.
// Compile-time layout => no per-store branching.
// ---------------------------------------------------------------------------
template <bool TRANSV>
__device__ __forceinline__ void qkv_panel(
        _Float16* Ws, const float* __restrict__ W, const float* __restrict__ bias,
        _Float16* __restrict__ outp, float oscale, const v16h* afrag,
        int tid, int wave, int g, int m, long rowblk) {
    __syncthreads();   // all waves done reading previous Ws contents
    for (int idx = tid; idx < 128 * 128 / 4; idx += 128) {
        int e = idx * 4; int r = e >> 7; int c = e & 127;
        float4 v = *(const float4*)(W + (long)r * 128 + c);
        _Float16* d = &Ws[r * 136 + c];
        d[0] = (_Float16)v.x; d[1] = (_Float16)v.y;
        d[2] = (_Float16)v.z; d[3] = (_Float16)v.w;
    }
    __syncthreads();

#pragma unroll
    for (int nt = 0; nt < 8; ++nt) {
        float bcol = bias[nt * 16 + m];
        v8f acc;
#pragma unroll
        for (int c = 0; c < 8; ++c) acc[c] = bcol;
#pragma unroll
        for (int kt = 0; kt < 4; ++kt) {
            const _Float16* bp = &Ws[(nt * 16 + m) * 136 + kt * 32 + 8 * g];
            v16h bfrag = ld_frag(bp, bp + 16);
            acc = __builtin_amdgcn_wmma_f32_16x16x32_f16(
                false, afrag[kt], false, bfrag, (short)0, acc, false, false);
        }
#pragma unroll
        for (int c = 0; c < 8; ++c) {
            long r = rowblk + wave * 16 + c + 8 * g;   // global row (b,t,n)
            int  e = nt * 16 + m;
            long b = r / ((long)Tq * NH);
            long t = (r / NH) % Tq;
            long n = r % NH;
            long head = b * NH + n;
            _Float16 h = (_Float16)(acc[c] * oscale);
            if (TRANSV)
                outp[(head * Dh + e) * (long)Tq + t] = h;
            else
                outp[(head * (long)Tq + t) * Dh + e] = h;
        }
    }
}

// ---------------------------------------------------------------------------
// Fused QKV projection: stages X tile + A-fragments once, three panels.
// Q is pre-scaled by 1/sqrt(D).  Block: 128 threads (4 waves), 64 rows.
// ---------------------------------------------------------------------------
__global__ __launch_bounds__(128, 1) void proj_qkv_kernel(
        const float* __restrict__ x,
        const float* __restrict__ wq, const float* __restrict__ bq,
        const float* __restrict__ wk, const float* __restrict__ bk,
        const float* __restrict__ wv, const float* __restrict__ bv,
        _Float16* __restrict__ Qh, _Float16* __restrict__ Kh,
        _Float16* __restrict__ Vt) {
    __shared__ __align__(16) _Float16 Xs[64 * 136];
    __shared__ __align__(16) _Float16 Ws[128 * 136];

    const int tid = threadIdx.x;                 // 128 threads
    const long rowblk = (long)blockIdx.x * 64;

    // Stage X tile (64 x 128) -> f16 LDS, K-contiguous (read x only once)
    for (int idx = tid; idx < 64 * 128 / 4; idx += 128) {
        int e = idx * 4; int r = e >> 7; int c = e & 127;
        float4 v = *(const float4*)(x + (rowblk + r) * 128 + c);
        _Float16* d = &Xs[r * 136 + c];
        d[0] = (_Float16)v.x; d[1] = (_Float16)v.y;
        d[2] = (_Float16)v.z; d[3] = (_Float16)v.w;
    }
    __syncthreads();

    const int wave = tid >> 5, lane = tid & 31, g = lane >> 4, m = lane & 15;

    // A fragments: wave's 16 rows, 4 K-tiles of 32 (loaded once, reused x3)
    v16h afrag[4];
#pragma unroll
    for (int kt = 0; kt < 4; ++kt) {
        const _Float16* p = &Xs[(wave * 16 + m) * 136 + kt * 32 + 8 * g];
        afrag[kt] = ld_frag(p, p + 16);
    }

    const float qscale = 0.08838834764831845f;   // 1/sqrt(128)
    qkv_panel<false>(Ws, wq, bq, Qh, qscale, afrag, tid, wave, g, m, rowblk);
    qkv_panel<false>(Ws, wk, bk, Kh, 1.0f,   afrag, tid, wave, g, m, rowblk);
    qkv_panel<true >(Ws, wv, bv, Vt, 1.0f,   afrag, tid, wave, g, m, rowblk);
}

// ---------------------------------------------------------------------------
// Output projection: out = att @ wp^T + bp, f16 att rows -> f32 out.
// ---------------------------------------------------------------------------
__global__ __launch_bounds__(128, 1) void proj_out_kernel(
        const _Float16* __restrict__ X,
        const float* __restrict__ W,
        const float* __restrict__ bias,
        float* __restrict__ outp) {
    __shared__ __align__(16) _Float16 Xs[64 * 136];
    __shared__ __align__(16) _Float16 Ws[128 * 136];

    const int tid = threadIdx.x;
    const long rowblk = (long)blockIdx.x * 64;

    for (int idx = tid; idx < 64 * 128 / 8; idx += 128) {
        int e = idx * 8; int r = e >> 7; int c = e & 127;
        *(v8h*)&Xs[r * 136 + c] = *(const v8h*)(X + (rowblk + r) * 128 + c);
    }
    for (int idx = tid; idx < 128 * 128 / 4; idx += 128) {
        int e = idx * 4; int r = e >> 7; int c = e & 127;
        float4 v = *(const float4*)(W + (long)r * 128 + c);
        _Float16* d = &Ws[r * 136 + c];
        d[0] = (_Float16)v.x; d[1] = (_Float16)v.y;
        d[2] = (_Float16)v.z; d[3] = (_Float16)v.w;
    }
    __syncthreads();

    const int wave = tid >> 5, lane = tid & 31, g = lane >> 4, m = lane & 15;

    v16h afrag[4];
#pragma unroll
    for (int kt = 0; kt < 4; ++kt) {
        const _Float16* p = &Xs[(wave * 16 + m) * 136 + kt * 32 + 8 * g];
        afrag[kt] = ld_frag(p, p + 16);
    }

#pragma unroll
    for (int nt = 0; nt < 8; ++nt) {
        float bcol = bias[nt * 16 + m];
        v8f acc;
#pragma unroll
        for (int c = 0; c < 8; ++c) acc[c] = bcol;
#pragma unroll
        for (int kt = 0; kt < 4; ++kt) {
            const _Float16* bp = &Ws[(nt * 16 + m) * 136 + kt * 32 + 8 * g];
            v16h bfrag = ld_frag(bp, bp + 16);
            acc = __builtin_amdgcn_wmma_f32_16x16x32_f16(
                false, afrag[kt], false, bfrag, (short)0, acc, false, false);
        }
#pragma unroll
        for (int c = 0; c < 8; ++c) {
            long r = rowblk + wave * 16 + c + 8 * g;
            outp[r * 128 + nt * 16 + m] = acc[c];
        }
    }
}

// ---------------------------------------------------------------------------
// Flash attention: grid (T/128, B*NH), 256 threads (8 waves).
// Wave w owns queries q0 = qb + 16w .. +15, full D accumulator (16x128 f32).
// Q pre-scaled by 1/sqrt(D).  64-key blocks: 32 WMMAs per softmax round.
// ---------------------------------------------------------------------------
__global__ __launch_bounds__(256, 1) void attn_kernel(
        const _Float16* __restrict__ Qh,
        const _Float16* __restrict__ Kh,
        const _Float16* __restrict__ Vth,
        _Float16* __restrict__ Att) {
    __shared__ __align__(16) _Float16 Kt[64 * 136];     // keys x d      (17.4 KB)
    __shared__ __align__(16) _Float16 Vs[128 * 72];     // d x keys      (18.4 KB)
    __shared__ __align__(16) _Float16 Ps[8 * 16 * 72];  // per-wave P    (18.4 KB)

    const int tid  = threadIdx.x;
    const int wave = tid >> 5, lane = tid & 31, g = lane >> 4, m = lane & 15;
    const int head = blockIdx.y;
    const int qb   = blockIdx.x * 128;
    const long hb  = (long)head * Tq * Dh;
    const _Float16* Qp = Qh + hb;
    const _Float16* Kp = Kh + hb;
    const _Float16* Vp = Vth + hb;     // [d][t]
    const int q0 = qb + wave * 16;

    // staging assignments (loop-invariant): 64 B per thread per tile
    const int kr = tid >> 2, kco = (tid & 3) * 32;   // K tile: 64 rows x 4 chunks
    const int vdr = tid >> 1, vco = (tid & 1) * 32;  // V^T tile: 128 rows x 2 chunks

    // Q fragments: held in registers for the whole kernel
    v16h qf[4];
#pragma unroll
    for (int kt = 0; kt < 4; ++kt) {
        const _Float16* p = Qp + (long)(q0 + m) * Dh + kt * 32 + 8 * g;
        qf[kt] = make16(*(const v8h*)p, *(const v8h*)(p + 16));
    }

    v8f acc[8];
    float mrun[8], lrun[8];
#pragma unroll
    for (int nt = 0; nt < 8; ++nt) acc[nt] = zero8();
#pragma unroll
    for (int c = 0; c < 8; ++c) { mrun[c] = -1e30f; lrun[c] = 0.0f; }

    const int nkb = (qb + 128) / 64;           // causal: only blocks <= diag

    for (int kblk = 0; kblk < nkb; ++kblk) {
        const int key0 = kblk * 64;
        // --- stage K tile: 64 keys x 128 d (16 KB) ---
        {
            const v8h* src = (const v8h*)(Kp + (long)(key0 + kr) * Dh + kco);
            v8h a = src[0], b = src[1], c2 = src[2], d2 = src[3];
            _Float16* dst = &Kt[kr * 136 + kco];
            *(v8h*)(dst)      = a;  *(v8h*)(dst + 8)  = b;
            *(v8h*)(dst + 16) = c2; *(v8h*)(dst + 24) = d2;
        }
        // --- stage V^T tile: 128 d x 64 keys (16 KB) ---
        {
            const v8h* src = (const v8h*)(Vp + (long)vdr * Tq + key0 + vco);
            v8h a = src[0], b = src[1], c2 = src[2], d2 = src[3];
            _Float16* dst = &Vs[vdr * 72 + vco];
            *(v8h*)(dst)      = a;  *(v8h*)(dst + 8)  = b;
            *(v8h*)(dst + 16) = c2; *(v8h*)(dst + 24) = d2;
        }
        // prefetch next tile into cache while we compute this one
        if (kblk + 1 < nkb) {
            __builtin_prefetch(Kp + (long)(key0 + 64 + kr) * Dh + kco, 0, 1);
            __builtin_prefetch(Vp + (long)vdr * Tq + key0 + 64 + vco, 0, 1);
        }
        __syncthreads();

        // --- S = Q @ K^T : four 16x16 subtiles over K=128 (4 wmma each) ---
        v8f s[4];
#pragma unroll
        for (int sub = 0; sub < 4; ++sub) s[sub] = zero8();
#pragma unroll
        for (int kt = 0; kt < 4; ++kt) {
#pragma unroll
            for (int sub = 0; sub < 4; ++sub) {
                const _Float16* bp = &Kt[(sub * 16 + m) * 136 + kt * 32 + 8 * g];
                v16h bf = ld_frag(bp, bp + 16);
                s[sub] = __builtin_amdgcn_wmma_f32_16x16x32_f16(
                    false, qf[kt], false, bf, (short)0, s[sub], false, false);
            }
        }

        // --- causal mask: only diagonal blocks need it (wave-uniform test) ---
        if (key0 + 63 > q0) {
#pragma unroll
            for (int c = 0; c < 8; ++c) {
                int q = q0 + c + 8 * g;
#pragma unroll
                for (int sub = 0; sub < 4; ++sub)
                    if (key0 + sub * 16 + m > q) s[sub][c] = -1e30f;
            }
        }

        // --- online softmax (per row c, reductions across 16-lane half) ---
#pragma unroll
        for (int c = 0; c < 8; ++c) {
            float mx = fmaxf(fmaxf(s[0][c], s[1][c]), fmaxf(s[2][c], s[3][c]));
#pragma unroll
            for (int off = 1; off < 16; off <<= 1)
                mx = fmaxf(mx, __shfl_xor(mx, off, 32));
            float mnew  = fmaxf(mrun[c], mx);
            float alpha = __expf(mrun[c] - mnew);
            float p0 = __expf(s[0][c] - mnew);
            float p1 = __expf(s[1][c] - mnew);
            float p2 = __expf(s[2][c] - mnew);
            float p3 = __expf(s[3][c] - mnew);
            float rs = (p0 + p1) + (p2 + p3);
#pragma unroll
            for (int off = 1; off < 16; off <<= 1)
                rs += __shfl_xor(rs, off, 32);
            lrun[c] = lrun[c] * alpha + rs;
            mrun[c] = mnew;
            s[0][c] = p0; s[1][c] = p1; s[2][c] = p2; s[3][c] = p3;
#pragma unroll
            for (int nt = 0; nt < 8; ++nt) acc[nt][c] *= alpha;
        }

        // --- C-layout P -> A-fragments via per-wave LDS (in-order per wave) ---
        _Float16* pb = &Ps[wave * 16 * 72];
#pragma unroll
        for (int c = 0; c < 8; ++c) {
            int row = (c + 8 * g) * 72;
#pragma unroll
            for (int sub = 0; sub < 4; ++sub)
                pb[row + sub * 16 + m] = (_Float16)s[sub][c];
        }
        const _Float16* pa = &pb[m * 72];
        v16h pf0 = make16(*(const v8h*)(pa + 8 * g),      *(const v8h*)(pa + 16 + 8 * g));
        v16h pf1 = make16(*(const v8h*)(pa + 32 + 8 * g), *(const v8h*)(pa + 48 + 8 * g));

        // --- O += P @ V : 8 n-tiles of d, 2 k-steps each ---
#pragma unroll
        for (int nt = 0; nt < 8; ++nt) {
            const _Float16* bp0 = &Vs[(nt * 16 + m) * 72 + 8 * g];
            const _Float16* bp1 = bp0 + 32;
            v16h bf0 = ld_frag(bp0, bp0 + 16);
            v16h bf1 = ld_frag(bp1, bp1 + 16);
            acc[nt] = __builtin_amdgcn_wmma_f32_16x16x32_f16(
                false, pf0, false, bf0, (short)0, acc[nt], false, false);
            acc[nt] = __builtin_amdgcn_wmma_f32_16x16x32_f16(
                false, pf1, false, bf1, (short)0, acc[nt], false, false);
        }
        __syncthreads();
    }

    // --- epilogue: normalize, store att as f16 in (b,t,n,d) row order ---
    int b = head / NH, n = head % NH;
#pragma unroll
    for (int c = 0; c < 8; ++c) {
        int t = q0 + c + 8 * g;
        long r = ((long)b * Tq + t) * NH + n;
        float inv = 1.0f / lrun[c];
#pragma unroll
        for (int nt = 0; nt < 8; ++nt)
            Att[r * Dh + nt * 16 + m] = (_Float16)(acc[nt][c] * inv);
    }
}

extern "C" void kernel_launch(void* const* d_in, const int* in_sizes, int n_in,
                              void* d_out, int out_size, void* d_ws, size_t ws_size,
                              hipStream_t stream) {
    const float* x  = (const float*)d_in[0];
    const float* wq = (const float*)d_in[1];
    const float* bq = (const float*)d_in[2];
    const float* wk = (const float*)d_in[3];
    const float* bk = (const float*)d_in[4];
    const float* wv = (const float*)d_in[5];
    const float* bv = (const float*)d_in[6];
    const float* wp = (const float*)d_in[7];
    const float* bp = (const float*)d_in[8];
    float* out = (float*)d_out;

    const long SZ = (long)NB * NH * Tq * Dh;   // 8,388,608 halves = 16 MB each
    _Float16* ws = (_Float16*)d_ws;
    _Float16* Qh = ws;
    _Float16* Kh = ws + SZ;
    _Float16* Vt = ws + 2 * SZ;
    _Float16* At = ws + 3 * SZ;

    dim3 pgrid((NB * Tq * NH) / 64);           // 1024 blocks
    dim3 pblk(128);
    proj_qkv_kernel<<<pgrid, pblk, 0, stream>>>(x, wq, bq, wk, bk, wv, bv, Qh, Kh, Vt);

    dim3 agrid(Tq / 128, NB * NH);             // (16, 32)
    attn_kernel<<<agrid, 256, 0, stream>>>(Qh, Kh, Vt, At);

    proj_out_kernel<<<pgrid, pblk, 0, stream>>>(At, wp, bp, out);
}
